// MHA_20177756356632
// MI455X (gfx1250) — matrix operand order repaired and checked by reading, not compiled, not run on hardware
//
#include <hip/hip_runtime.h>
#include <hip/hip_bf16.h>

typedef _Float16 h16;
typedef __attribute__((ext_vector_type(16))) _Float16 v16h;
typedef __attribute__((ext_vector_type(8)))  float    v8f;

#define B_   4
#define L_   1024
#define C_   512
#define H_   8
#define DK_  64
#define DV_  192
#define HDV_ (H_*DV_)
#define BH_  (B_*H_)
#define RROWS_ 2048   // 2L-1 = 2047 real rows + 1 zero pad row

union V16U { v16h v; uint4 u[2]; };

__device__ __forceinline__ v8f wmma16(v16h a, v16h b, v8f c) {
  // D = A(16x32 f16) * B(32x16 f16) + C(16x16 f32)
  return __builtin_amdgcn_wmma_f32_16x16x32_f16(false, a, false, b, (short)0, c, false, false);
}

// A operand (row-major source, 16x32 tile): lane's halves are
// [8*half .. 8*half+7] then [16+8*half .. 23+8*half]  -> two b128 loads.
__device__ __forceinline__ v16h ldA(const h16* rowk0, int half) {
  V16U r;
  r.u[0] = *(const uint4*)(rowk0 + 8 * half);
  r.u[1] = *(const uint4*)(rowk0 + 16 + 8 * half);
  return r.v;
}
// B operand (lane = column, K contiguous per lane): halves
// [8*half .. 8*half+7] then [16+8*half .. 23+8*half] -> two b128 loads.
__device__ __forceinline__ v16h ldB(const h16* colk0, int half) {
  const h16* p = colk0 + 8 * half;
  V16U r;
  r.u[0] = *(const uint4*)p;
  r.u[1] = *(const uint4*)(p + 16);
  return r.v;
}

// ---------------------------------------------------------------------------
// fp32 -> f16 bulk convert (x)
// ---------------------------------------------------------------------------
__global__ void cvt16_kernel(const float* __restrict__ in, h16* __restrict__ out, int n) {
  int i = blockIdx.x * blockDim.x + threadIdx.x;
  if (i < n) out[i] = (h16)in[i];
}

// fp32 W[k][n] -> f16 Wt[n][k]  (so GEMM B operand is per-lane K-contiguous)
__global__ void wt_kernel(const float* __restrict__ W, h16* __restrict__ Wt, int K, int N) {
  int i = blockIdx.x * blockDim.x + threadIdx.x;
  if (i >= K * N) return;
  int n = i / K, k = i % K;
  Wt[(size_t)n * K + k] = (h16)W[(size_t)k * N + n];
}

// ---------------------------------------------------------------------------
// R = sinusoid_basis(2L-1, 192) @ We  -> Rh f16 [2048][64] (row 2047 zeroed)
// t4[h][r] = v[h] . R[r]              -> f32 [8][2048]
// ---------------------------------------------------------------------------
__global__ void prep_R_t4_kernel(const float* __restrict__ We, const float* __restrict__ vv,
                                 h16* __restrict__ Rh, float* __restrict__ t4) {
  const int r = blockIdx.x;    // 0..2047
  const int d = threadIdx.x;   // 0..63
  __shared__ float rowf[64];
  float val = 0.f;
  if (r < 2 * L_ - 1) {
    const float dist = (float)r - (float)(L_ - 1);
    const float cfr = 9.210340371976184f / 96.0f;  // ln(10000)/(NF/2)
    for (int f = 0; f < 96; ++f) {
      float ang = dist * expf(-cfr * (float)f);
      val += sinf(ang) * We[f * DK_ + d] + cosf(ang) * We[(f + 96) * DK_ + d];
    }
  }
  Rh[(size_t)r * DK_ + d] = (h16)val;
  rowf[d] = val;
  __syncthreads();
  if (d < H_) {
    float s = 0.f;
    for (int k = 0; k < DK_; ++k) s += vv[d * DK_ + k] * rowf[k];
    t4[d * RROWS_ + r] = s;
  }
}

// ---------------------------------------------------------------------------
// Projection GEMM (all f16 operands): out = Xh[4096,512] @ WtT + bias,
// head-split store. One wave computes a 16x64 tile (4 WMMA accumulators).
// TSTORE=0: [bh][i][D]; TSTORE=1: [bh][D][i] (transposed for V).
// ---------------------------------------------------------------------------
template <int N, int D, int TSTORE>
__global__ void __launch_bounds__(32)
proj_kernel(const h16* __restrict__ Xh, const h16* __restrict__ Wt,
            const float* __restrict__ bias, h16* __restrict__ out) {
  const int lane = threadIdx.x & 31;
  const int n16 = lane & 15, half = lane >> 4;
  const int row0 = blockIdx.x * 16;
  const int n0 = blockIdx.y * 64;
  v8f acc[4];
#pragma unroll
  for (int t = 0; t < 4; ++t) acc[t] = v8f{};
  const h16* ar = Xh + (size_t)(row0 + n16) * C_;
  for (int k0 = 0; k0 < C_; k0 += 32) {
    v16h a = ldA(ar + k0, half);
#pragma unroll
    for (int t = 0; t < 4; ++t) {
      v16h b = ldB(Wt + (size_t)(n0 + 16 * t + n16) * C_ + k0, half);
      acc[t] = wmma16(a, b, acc[t]);
    }
  }
#pragma unroll
  for (int t = 0; t < 4; ++t) {
    const int col = n0 + 16 * t + n16;
    const float bval = bias[col];
    const int hh = col / D, d = col % D;   // compile-time D -> no div
#pragma unroll
    for (int r = 0; r < 8; ++r) {
      int mm = r + 8 * half;
      int row = row0 + mm;
      int bb = row >> 10, ii = row & (L_ - 1);
      int bh = bb * H_ + hh;
      size_t idx = TSTORE ? ((size_t)bh * D + d) * L_ + ii
                          : ((size_t)bh * L_ + ii) * D + d;
      out[idx] = (h16)(acc[t][r] + bval);
    }
  }
}

// ---------------------------------------------------------------------------
// t3[bh*L + j] = u[h] . K[bh][j]
// ---------------------------------------------------------------------------
__global__ void t3_kernel(const h16* __restrict__ Kh, const float* __restrict__ u,
                          float* __restrict__ t3) {
  int g = blockIdx.x * blockDim.x + threadIdx.x;
  if (g >= BH_ * L_) return;
  int hh = (g >> 10) & (H_ - 1);
  const h16* kr = Kh + (size_t)g * DK_;
  float s = 0.f;
  for (int d = 0; d < DK_; ++d) s += u[hh * DK_ + d] * (float)kr[d];
  t3[g] = s;
}

// ---------------------------------------------------------------------------
// Attention: one wave per (bh, 16-row block). Scores (term1..4) -> LDS f16,
// row softmax stats, then P @ V with 12 f32 WMMA accumulators.
// ---------------------------------------------------------------------------
__global__ void __launch_bounds__(32)
attn_kernel(const h16* __restrict__ Qh, const h16* __restrict__ Kh,
            const h16* __restrict__ Vt, const h16* __restrict__ Rh,
            const float* __restrict__ t3, const float* __restrict__ t4g,
            h16* __restrict__ Obuf) {
  __shared__ h16   sc[16][L_];     // 32 KB score scratch
  __shared__ float gbuf[16][32];   // rel-position strip G
  __shared__ float t4buf[32];
  __shared__ float rowmax[16];
  __shared__ float rowsum[16];

  const int lane = threadIdx.x & 31;
  const int n16 = lane & 15, half = lane >> 4;
  const int i0 = blockIdx.x * 16;
  const int bh = blockIdx.y;
  const int b = bh >> 3, hh = bh & (H_ - 1);

  // Q A-tiles for both 32-wide K chunks (reused across all j tiles)
  const h16* qr = Qh + ((size_t)bh * L_ + i0 + n16) * DK_;
  v16h qa0 = ldA(qr, half);
  v16h qa1 = ldA(qr + 32, half);

  // ---- Phase A: logits into LDS ----
  for (int j0 = 0; j0 < L_; j0 += 16) {
    const h16* kr = Kh + ((size_t)bh * L_ + j0 + n16) * DK_;
    if (j0 + 16 < L_)  // prefetch next K tile (global_prefetch_b8)
      __builtin_prefetch(kr + 16 * DK_, 0, 3);
    v8f acc = {};
    acc = wmma16(qa0, ldB(kr, half), acc);
    acc = wmma16(qa1, ldB(kr + 32, half), acc);

    const int c = (L_ - 1) + j0 - i0;  // strip center index into R
    // term2 strip: G[m][s] = Q[i0+m] . R[c-15+s], s=0..31
#pragma unroll
    for (int t = 0; t < 2; ++t) {
      const h16* rr = Rh + (size_t)(c - 15 + 16 * t + n16) * DK_;
      v8f g = {};
      g = wmma16(qa0, ldB(rr, half), g);
      g = wmma16(qa1, ldB(rr + 32, half), g);
#pragma unroll
      for (int r = 0; r < 8; ++r) gbuf[r + 8 * half][16 * t + n16] = g[r];
    }
    t4buf[lane] = t4g[hh * RROWS_ + (c - 15) + lane];
    const float t3v = t3[(size_t)bh * L_ + j0 + n16];
    __syncthreads();
#pragma unroll
    for (int r = 0; r < 8; ++r) {
      int mm = r + 8 * half;
      int srel = n16 - mm + 15;                 // 0..30, always in range
      float s = acc[r] + gbuf[mm][srel] + t3v + t4buf[srel];
      sc[mm][j0 + n16] = (h16)s;
    }
    __syncthreads();
  }

  // ---- Phase B: per-row max & sum (lane handles row n16, half the columns) ----
  {
    const int m = n16;
    float mx = -1e30f;
    for (int j = 512 * half; j < 512 * half + 512; ++j)
      mx = fmaxf(mx, (float)sc[m][j]);
    mx = fmaxf(mx, __shfl_xor(mx, 16));
    if (half == 0) rowmax[m] = mx;
    float sm = 0.f;
    for (int j = 512 * half; j < 512 * half + 512; ++j)
      sm += __expf((float)sc[m][j] - mx);
    sm += __shfl_xor(sm, 16);
    if (half == 0) rowsum[m] = sm;
  }
  __syncthreads();

  // ---- Phase C: O = softmax(S) @ V ----
  v8f o[12];
#pragma unroll
  for (int t = 0; t < 12; ++t) o[t] = v8f{};
  const float rm = rowmax[n16];  // A-layout row for this lane
  const h16* vbase = Vt + (size_t)bh * DV_ * L_;
  for (int j0 = 0; j0 < L_; j0 += 32) {
    if (j0 + 32 < L_)  // prefetch next V stripe
      __builtin_prefetch(vbase + (size_t)n16 * L_ + j0 + 32, 0, 3);
    v16h pa;
#pragma unroll
    for (int v = 0; v < 8; ++v) {
      int kb = 2 * v + ((v >= 4) ? 8 : 0) + 8 * half;
      pa[2 * v]     = (h16)__expf((float)sc[n16][j0 + kb]     - rm);
      pa[2 * v + 1] = (h16)__expf((float)sc[n16][j0 + kb + 1] - rm);
    }
#pragma unroll
    for (int t = 0; t < 12; ++t) {
      // Vt is [bh][dv][L] so B operand is per-lane contiguous in K=j
      const h16* vr = vbase + (size_t)(t * 16 + n16) * L_ + j0;
      o[t] = wmma16(pa, ldB(vr, half), o[t]);
    }
  }
#pragma unroll
  for (int r = 0; r < 8; ++r) {
    int mm = r + 8 * half;
    float inv = 1.0f / rowsum[mm];
    size_t row = (size_t)b * L_ + i0 + mm;
#pragma unroll
    for (int t = 0; t < 12; ++t)
      Obuf[row * HDV_ + hh * DV_ + t * 16 + n16] = (h16)(o[t][r] * inv);
  }
}

// ---------------------------------------------------------------------------
// Output GEMM: out(f32) = Obuf[4096,1536](f16) @ WoT[512][1536](f16) + bo
// One wave computes a 16x64 tile.
// ---------------------------------------------------------------------------
__global__ void __launch_bounds__(32)
outproj_kernel(const h16* __restrict__ A, const h16* __restrict__ WoT,
               const float* __restrict__ bo, float* __restrict__ out) {
  const int lane = threadIdx.x & 31;
  const int n16 = lane & 15, half = lane >> 4;
  const int row0 = blockIdx.x * 16;
  const int n0 = blockIdx.y * 64;
  v8f acc[4];
#pragma unroll
  for (int t = 0; t < 4; ++t) acc[t] = v8f{};
  const h16* ar = A + (size_t)(row0 + n16) * HDV_;
  for (int k0 = 0; k0 < HDV_; k0 += 32) {
    v16h a = ldA(ar + k0, half);
#pragma unroll
    for (int t = 0; t < 4; ++t) {
      v16h b = ldB(WoT + (size_t)(n0 + 16 * t + n16) * HDV_ + k0, half);
      acc[t] = wmma16(a, b, acc[t]);
    }
  }
#pragma unroll
  for (int t = 0; t < 4; ++t) {
    const float bval = bo[n0 + 16 * t + n16];
#pragma unroll
    for (int r = 0; r < 8; ++r) {
      size_t row = row0 + r + 8 * half;
      out[row * C_ + n0 + 16 * t + n16] = acc[t][r] + bval;
    }
  }
}

// ---------------------------------------------------------------------------
extern "C" void kernel_launch(void* const* d_in, const int* in_sizes, int n_in,
                              void* d_out, int out_size, void* d_ws, size_t ws_size,
                              hipStream_t stream) {
  (void)in_sizes; (void)n_in; (void)out_size; (void)ws_size;
  const float* x  = (const float*)d_in[0];
  const float* Wq = (const float*)d_in[1];
  const float* bq = (const float*)d_in[2];
  const float* Wk = (const float*)d_in[3];
  const float* bk = (const float*)d_in[4];
  const float* Wv = (const float*)d_in[5];
  const float* bv = (const float*)d_in[6];
  const float* Wo = (const float*)d_in[7];
  const float* bo = (const float*)d_in[8];
  const float* We = (const float*)d_in[9];
  const float* u  = (const float*)d_in[10];
  const float* v  = (const float*)d_in[11];
  float* out = (float*)d_out;

  // workspace layout (~42 MB total)
  char* ws = (char*)d_ws;
  size_t off = 0;
  h16*   Qh   = (h16*)(ws + off);   off += (size_t)BH_ * L_ * DK_ * 2;   // 4 MB
  h16*   Kh   = (h16*)(ws + off);   off += (size_t)BH_ * L_ * DK_ * 2;   // 4 MB
  h16*   Vt   = (h16*)(ws + off);   off += (size_t)BH_ * DV_ * L_ * 2;   // 12 MB (transposed)
  h16*   Rh   = (h16*)(ws + off);   off += (size_t)RROWS_ * DK_ * 2;     // 256 KB
  float* t4b  = (float*)(ws + off); off += (size_t)H_ * RROWS_ * 4;      // 64 KB
  float* t3b  = (float*)(ws + off); off += (size_t)BH_ * L_ * 4;         // 128 KB
  h16*   Obuf = (h16*)(ws + off);   off += (size_t)B_ * L_ * HDV_ * 2;   // 12 MB
  h16*   Xh   = (h16*)(ws + off);   off += (size_t)B_ * L_ * C_ * 2;     // 4 MB
  h16*   WqT  = (h16*)(ws + off);   off += (size_t)C_ * C_ * 2;          // 512 KB
  h16*   WkT  = (h16*)(ws + off);   off += (size_t)C_ * C_ * 2;          // 512 KB
  h16*   WvT  = (h16*)(ws + off);   off += (size_t)C_ * HDV_ * 2;        // 1.5 MB
  h16*   WoT  = (h16*)(ws + off);   off += (size_t)HDV_ * C_ * 2;        // 1.5 MB

  // staging: convert / transpose to f16
  cvt16_kernel<<<dim3((B_ * L_ * C_) / 256), dim3(256), 0, stream>>>(x, Xh, B_ * L_ * C_);
  wt_kernel<<<dim3((C_ * C_) / 256), dim3(256), 0, stream>>>(Wq, WqT, C_, C_);
  wt_kernel<<<dim3((C_ * C_) / 256), dim3(256), 0, stream>>>(Wk, WkT, C_, C_);
  wt_kernel<<<dim3((C_ * HDV_) / 256), dim3(256), 0, stream>>>(Wv, WvT, C_, HDV_);
  wt_kernel<<<dim3((HDV_ * C_) / 256), dim3(256), 0, stream>>>(Wo, WoT, HDV_, C_);
  prep_R_t4_kernel<<<dim3(RROWS_), dim3(64), 0, stream>>>(We, v, Rh, t4b);

  proj_kernel<512, DK_, 0><<<dim3(256, 8), dim3(32), 0, stream>>>(Xh, WqT, bq, Qh);
  proj_kernel<512, DK_, 0><<<dim3(256, 8), dim3(32), 0, stream>>>(Xh, WkT, bk, Kh);
  proj_kernel<HDV_, DV_, 1><<<dim3(256, 24), dim3(32), 0, stream>>>(Xh, WvT, bv, Vt);
  t3_kernel<<<dim3((BH_ * L_) / 256), dim3(256), 0, stream>>>(Kh, u, t3b);
  attn_kernel<<<dim3(L_ / 16, BH_), dim3(32), 0, stream>>>(Qh, Kh, Vt, Rh, t3b, t4b, Obuf);
  outproj_kernel<<<dim3(256, 8), dim3(32), 0, stream>>>(Obuf, WoT, bo, out);
}
